// Tissue_GCN_cls_3624952398638
// MI455X (gfx1250) — matrix operand order repaired and checked by reading, not compiled, not run on hardware
//
#include <hip/hip_runtime.h>
#include <hip/hip_bf16.h>
#include <stdint.h>

#define NN   50000
#define EE   640000
#define HID  128
#define EXPD 256
#define MEPS 1e-7f
#define LNEPS 1e-5f

typedef __attribute__((ext_vector_type(16))) __bf16          v16bf;
typedef __attribute__((ext_vector_type(8)))  float           v8f;
typedef __attribute__((ext_vector_type(8)))  unsigned short  v8u;
typedef __attribute__((ext_vector_type(16))) unsigned short  v16u;
typedef __attribute__((ext_vector_type(4)))  unsigned int    v4u32;
typedef __attribute__((ext_vector_type(8)))  int             v8i32;
typedef __attribute__((ext_vector_type(4)))  int             v4i32;

// ---------- helpers ----------
__device__ __forceinline__ unsigned short f2bf(float f) {
    unsigned u = __float_as_uint(f);
    u += 0x7fffu + ((u >> 16) & 1u);          // round-to-nearest-even
    return (unsigned short)(u >> 16);
}
__device__ __forceinline__ unsigned pk2bf(float a, float b) {
    return (unsigned)f2bf(a) | ((unsigned)f2bf(b) << 16);
}
__device__ __forceinline__ v16bf mkfrag(v8u lo, v8u hi) {
    v16u v = __builtin_shufflevector(lo, hi, 0,1,2,3,4,5,6,7,8,9,10,11,12,13,14,15);
    return __builtin_bit_cast(v16bf, v);
}
__device__ __forceinline__ v8f bwmma(v16bf a, v16bf b, v8f c) {
    return __builtin_amdgcn_wmma_f32_16x16x32_bf16(false, a, false, b, (short)0, c,
                                                   false, false);
}
// A fragment from bf16-bit LDS tile, row stride `stride` elements.
// ISA 16-bit A layout: lanes0-15 row m hold K {0..7,16..23}+kbase,
// lanes16-31 hold K {8..15,24..31}+kbase.
__device__ __forceinline__ v16bf fragA(const unsigned short* s, int stride,
                                       int ln, int laneHi, int kbase) {
    const unsigned short* p = s + ln * stride + kbase + laneHi * 8;
    v8u lo = *(const v8u*)p;
    v8u hi = *(const v8u*)(p + 16);
    return mkfrag(lo, hi);
}
// A fragment converted on the fly from an fp32 LDS image (TDM-loaded).
__device__ __forceinline__ v16bf fragAf(const float* s, int stride,
                                        int ln, int laneHi, int kbase) {
    const float* p = s + ln * stride + kbase + laneHi * 8;
    v16u r;
#pragma unroll
    for (int i = 0; i < 8; ++i) { r[i] = f2bf(p[i]); r[8 + i] = f2bf(p[16 + i]); }
    return __builtin_bit_cast(v16bf, r);
}
// B fragment from K-transposed slab sBT[n][32]: lane column n holds
// K = laneHi*8 + {0..7, 16..23} (same order as A).
__device__ __forceinline__ v16bf fragB(const unsigned short* s, int nrow, int laneHi) {
    const unsigned short* p = s + nrow * 32 + laneHi * 8;
    v8u lo = *(const v8u*)p;
    v8u hi = *(const v8u*)(p + 16);
    return mkfrag(lo, hi);
}
// order-preserving float<->uint key for atomic float max
__device__ __forceinline__ unsigned fkey(float f) {
    unsigned u = __float_as_uint(f);
    return (u & 0x80000000u) ? ~u : (u | 0x80000000u);
}
__device__ __forceinline__ float funkey(unsigned u) {
    return (u & 0x80000000u) ? __uint_as_float(u & 0x7fffffffu)
                             : __uint_as_float(~u);
}
// TDM: DMA a [rows x cols] fp32 tile (row stride `gstride` elems) into LDS.
// Issued once per calling wave; tracked by TENSORcnt.
__device__ __forceinline__ void tdm_load_f32(const float* gaddr, unsigned lds_byte,
                                             unsigned cols, unsigned rows,
                                             unsigned long long gstride,
                                             unsigned long long tdim0,
                                             unsigned long long tdim1) {
    unsigned long long ga = (unsigned long long)(uintptr_t)gaddr;
    v4u32 g0;
    g0[0] = 1u;                                   // count=1, no gather/restore
    g0[1] = lds_byte;                             // lds_addr
    g0[2] = (unsigned)ga;                         // global_addr[31:0]
    g0[3] = ((unsigned)(ga >> 32) & 0x1ffffffu) | (2u << 30);  // addr[56:32]|type=2
    // group1: wg_mask=0, data_size=2 (4B), no barrier/iterate/pad
    unsigned long long q0 = (2ull << 16) | ((tdim0 & 0xffffull) << 48);
    unsigned long long q1 = (tdim0 >> 16) | (tdim1 << 16) |
                            ((unsigned long long)cols << 48);      // tile_dim0
    unsigned long long q2 = (unsigned long long)rows |             // tile_dim1
                            ((gstride & 0xffffffffull) << 32);     // dim0_stride lo
    unsigned long long q3 = (gstride >> 32);                       // dim0_stride hi
    v8i32 g1;
    g1[0] = (int)q0; g1[1] = (int)(q0 >> 32);
    g1[2] = (int)q1; g1[3] = (int)(q1 >> 32);
    g1[4] = (int)q2; g1[5] = (int)(q2 >> 32);
    g1[6] = (int)q3; g1[7] = (int)(q3 >> 32);
    v4i32 z4 = {0, 0, 0, 0};
    v8i32 z8 = {0, 0, 0, 0, 0, 0, 0, 0};
    __builtin_amdgcn_tensor_load_to_lds(g0, g1, z4, z4, z8, 0);
}

// ---------- 1) FC: h0 = relu(x @ fc_w + fc_b), write out[:,0:128] ----------
__global__ __launch_bounds__(256)
void k_fc(const float* __restrict__ x, const float* __restrict__ w,
          const float* __restrict__ b, float* __restrict__ out) {
    __shared__ float sXf[16 * 512];            // TDM-loaded A image (fp32)
    __shared__ unsigned short sBT[128 * 32];   // [n][k] bf16 bits (transposed)
    const int tid = threadIdx.x;
    const int wave = tid >> 5, lane = tid & 31;
    const int laneHi = lane >> 4, ln = lane & 15;
    const int row0 = blockIdx.x * 16;

    v8f acc = {};
    for (int half = 0; half < 2; ++half) {
        if (wave == 0) {   // one TDM op per wave: 16x512 fp32 tile -> LDS
            tdm_load_f32(x + (size_t)row0 * 1024 + half * 512,
                         (unsigned)(uintptr_t)sXf,
                         /*cols=*/512, /*rows=*/16, /*gstride=*/1024,
                         /*tdim0=*/1024, /*tdim1=*/16);
            __builtin_amdgcn_s_wait_tensorcnt(0);
        }
        __syncthreads();
        for (int k0 = 0; k0 < 512; k0 += 32) {
            const int kg = half * 512 + k0;
            // stage B tile transposed, packed pairs: 32x128 -> sBT[n][kk]
            for (int j = 0; j < 8; ++j) {
                int idx = j * 256 + tid;            // pair idx [0,2048)
                int n = idx & 127, kp = idx >> 7;   // coalesced over n
                float va = w[(size_t)(kg + 2 * kp) * 128 + n];
                float vb = w[(size_t)(kg + 2 * kp + 1) * 128 + n];
                *(unsigned*)&sBT[n * 32 + 2 * kp] = pk2bf(va, vb);
            }
            __syncthreads();
            v16bf af = fragAf(sXf, 512, ln, laneHi, k0);
            v16bf bf = fragB(sBT, wave * 16 + ln, laneHi);
            acc = bwmma(af, bf, acc);
            __syncthreads();
        }
    }
    const int n = wave * 16 + ln;
    const float bias = b[n];
    for (int r = 0; r < 8; ++r) {
        int m = r + laneHi * 8;                    // C/D layout
        float v = fmaxf(acc[r] + bias, 0.f);
        out[(size_t)(row0 + m) * 512 + n] = v;
    }
}

// ---------- 2) per-layer accumulator init ----------
__global__ __launch_bounds__(256)
void k_init(unsigned* __restrict__ mkey, float* __restrict__ den,
            float* __restrict__ num) {
    size_t i = (size_t)blockIdx.x * 256 + threadIdx.x;
    if (i < (size_t)NN * HID) { mkey[i] = 0u; den[i] = 0.f; num[i] = 0.f; }
}

// ---------- 3) edge pass A: segment max of s = t*(relu(h_src)+eps) ----------
__global__ __launch_bounds__(256)
void k_edge_max(const float* __restrict__ h, const long long* __restrict__ src,
                const long long* __restrict__ dst, const float* __restrict__ t,
                int layer, unsigned* __restrict__ mkey) {
    int tid = blockIdx.x * 256 + threadIdx.x;
    int e = tid >> 5, lane = tid & 31;
    if (e >= EE) return;
    const long long s = src[e], d = dst[e];
    const float tl = t[layer];
    const float4 hv = *(const float4*)(h + (size_t)s * 512 + lane * 4);
    unsigned* mk = mkey + (size_t)d * HID + lane * 4;
    atomicMax(mk + 0, fkey(tl * (fmaxf(hv.x, 0.f) + MEPS)));
    atomicMax(mk + 1, fkey(tl * (fmaxf(hv.y, 0.f) + MEPS)));
    atomicMax(mk + 2, fkey(tl * (fmaxf(hv.z, 0.f) + MEPS)));
    atomicMax(mk + 3, fkey(tl * (fmaxf(hv.w, 0.f) + MEPS)));
}

// ---------- 4) edge pass B: den += e^(s-m), num += msg*e^(s-m) ----------
__global__ __launch_bounds__(256)
void k_edge_sum(const float* __restrict__ h, const long long* __restrict__ src,
                const long long* __restrict__ dst, const float* __restrict__ t,
                int layer, const unsigned* __restrict__ mkey,
                float* __restrict__ den, float* __restrict__ num) {
    int tid = blockIdx.x * 256 + threadIdx.x;
    int e = tid >> 5, lane = tid & 31;
    if (e >= EE) return;
    const long long s = src[e], d = dst[e];
    const float tl = t[layer];
    const float4 hv = *(const float4*)(h + (size_t)s * 512 + lane * 4);
    const size_t o = (size_t)d * HID + lane * 4;
    float m[4] = {hv.x, hv.y, hv.z, hv.w};
#pragma unroll
    for (int c = 0; c < 4; ++c) {
        float msg = fmaxf(m[c], 0.f) + MEPS;
        float ex = __expf(tl * msg - funkey(mkey[o + c]));
        atomicAdd(den + o + c, ex);
        atomicAdd(num + o + c, msg * ex);
    }
}

// ---------- 5) node MLP (GEMM1 -> LN -> ReLU -> GEMM2 -> epilogue) ----------
__global__ __launch_bounds__(256)
void k_node_mlp(int layer, float* out,
                const float* __restrict__ w1, const float* __restrict__ b1,
                const float* __restrict__ g1, const float* __restrict__ be1,
                const float* __restrict__ w2, const float* __restrict__ b2,
                const float* __restrict__ ng, const float* __restrict__ nb,
                const float* __restrict__ num, const float* __restrict__ den) {
    __shared__ unsigned short sA[16 * 128];     // MLP input, bf16 bits
    __shared__ unsigned short sWT[256 * 32];    // weight slab, K-transposed
    __shared__ float          sU[16 * 256];     // GEMM1 out / LN input
    __shared__ unsigned short sUb[16 * 256];    // LN'ed bf16 for GEMM2
    __shared__ float          sC[16 * 128];     // GEMM2 out
    __shared__ float          sMu[16], sRs[16];

    const int tid = threadIdx.x;
    const int wave = tid >> 5, lane = tid & 31;
    const int laneHi = lane >> 4, ln = lane & 15;
    const int row0 = blockIdx.x * 16;

    const float* w1l = w1 + (size_t)layer * 128 * 256;
    const float* b1l = b1 + (size_t)layer * 256;
    const float* g1l = g1 + (size_t)layer * 256;
    const float* be1l = be1 + (size_t)layer * 256;
    const float* w2l = w2 + (size_t)layer * 256 * 128;
    const float* b2l = b2 + (size_t)layer * 128;
    const float* ngl = ng + (size_t)layer * 128;
    const float* nbl = nb + (size_t)layer * 128;
    const float* hprev = out + (size_t)layer * HID;        // col slice of out
    float* hnext = out + (size_t)(layer + 1) * HID;

    // phase 0: hin = aggr + h_prev  (aggr = num/den, 0 for empty segments)
    for (int j = 0; j < 4; ++j) {
        int idx = j * 256 + tid;                // pair idx [0,1024)
        int m = idx >> 6, c = (idx & 63) * 2;
        size_t o = (size_t)(row0 + m) * HID + c;
        size_t ho = (size_t)(row0 + m) * 512 + c;
        float d0 = den[o], d1 = den[o + 1];
        float a0 = d0 > 0.f ? num[o] / d0 : 0.f;
        float a1 = d1 > 0.f ? num[o + 1] / d1 : 0.f;
        *(unsigned*)&sA[m * 128 + c] = pk2bf(a0 + hprev[ho], a1 + hprev[ho + 1]);
    }
    __syncthreads();

    // GEMM1: [16,128] @ w1[128,256]; wave handles N-tiles {wave, wave+8}
    v8f acc0 = {}, acc1 = {};
    for (int kt = 0; kt < 4; ++kt) {
        for (int j = 0; j < 16; ++j) {          // 32x256 slab, packed pairs
            int idx = j * 256 + tid;            // pair idx [0,4096)
            int n = idx & 255, kp = idx >> 8;
            float va = w1l[(size_t)(kt * 32 + 2 * kp) * 256 + n];
            float vb = w1l[(size_t)(kt * 32 + 2 * kp + 1) * 256 + n];
            *(unsigned*)&sWT[n * 32 + 2 * kp] = pk2bf(va, vb);
        }
        __syncthreads();
        v16bf af = fragA(sA, 128, ln, laneHi, kt * 32);
        acc0 = bwmma(af, fragB(sWT, wave * 16 + ln, laneHi), acc0);
        acc1 = bwmma(af, fragB(sWT, (wave + 8) * 16 + ln, laneHi), acc1);
        __syncthreads();
    }
    {
        int n0 = wave * 16 + ln, n1 = (wave + 8) * 16 + ln;
        float bb0 = b1l[n0], bb1 = b1l[n1];
        for (int r = 0; r < 8; ++r) {
            int m = r + laneHi * 8;
            sU[m * 256 + n0] = acc0[r] + bb0;
            sU[m * 256 + n1] = acc1[r] + bb1;
        }
    }
    __syncthreads();

    // LN(256) + ReLU -> bf16
    if (tid < 16) {
        float s = 0.f, ss = 0.f;
        for (int c = 0; c < 256; ++c) { float v = sU[tid * 256 + c]; s += v; ss += v * v; }
        float mu = s * (1.f / 256.f);
        float var = ss * (1.f / 256.f) - mu * mu;
        sMu[tid] = mu; sRs[tid] = rsqrtf(var + LNEPS);
    }
    __syncthreads();
    for (int j = 0; j < 8; ++j) {
        int idx = j * 256 + tid;                // pair idx [0,2048)
        int m = idx >> 7, c = (idx & 127) * 2;
        float v0 = (sU[m * 256 + c] - sMu[m]) * sRs[m] * g1l[c] + be1l[c];
        float v1 = (sU[m * 256 + c + 1] - sMu[m]) * sRs[m] * g1l[c + 1] + be1l[c + 1];
        *(unsigned*)&sUb[m * 256 + c] = pk2bf(fmaxf(v0, 0.f), fmaxf(v1, 0.f));
    }
    __syncthreads();

    // GEMM2: [16,256] @ w2[256,128]; wave handles N-tile `wave`
    v8f acc = {};
    for (int kt = 0; kt < 8; ++kt) {
        for (int j = 0; j < 8; ++j) {           // 32x128 slab, packed pairs
            int idx = j * 256 + tid;            // pair idx [0,2048)
            int n = idx & 127, kp = idx >> 7;
            float va = w2l[(size_t)(kt * 32 + 2 * kp) * 128 + n];
            float vb = w2l[(size_t)(kt * 32 + 2 * kp + 1) * 128 + n];
            *(unsigned*)&sWT[n * 32 + 2 * kp] = pk2bf(va, vb);
        }
        __syncthreads();
        v16bf af = fragA(sUb, 256, ln, laneHi, kt * 32);
        acc = bwmma(af, fragB(sWT, wave * 16 + ln, laneHi), acc);
        __syncthreads();
    }
    {
        int n = wave * 16 + ln;
        float bb = b2l[n];
        for (int r = 0; r < 8; ++r) {
            int m = r + laneHi * 8;
            sC[m * 128 + n] = acc[r] + bb;
        }
    }
    __syncthreads();

    // epilogue: layer 0 -> raw conv out; layers 1,2 -> h + relu(LN(conv))
    if (layer != 0 && tid < 16) {
        float s = 0.f, ss = 0.f;
        for (int c = 0; c < 128; ++c) { float v = sC[tid * 128 + c]; s += v; ss += v * v; }
        float mu = s * (1.f / 128.f);
        float var = ss * (1.f / 128.f) - mu * mu;
        sMu[tid] = mu; sRs[tid] = rsqrtf(var + LNEPS);
    }
    __syncthreads();
    for (int j = 0; j < 8; ++j) {
        int idx = j * 256 + tid;
        int m = idx >> 7, c = idx & 127;
        float v = sC[m * 128 + c];
        if (layer != 0) {
            float cn = fmaxf((v - sMu[m]) * sRs[m] * ngl[c] + nbl[c], 0.f);
            v = hprev[(size_t)(row0 + m) * 512 + c] + cn;
        }
        hnext[(size_t)(row0 + m) * 512 + c] = v;
    }
}

// ---------- host ----------
extern "C" void kernel_launch(void* const* d_in, const int* in_sizes, int n_in,
                              void* d_out, int out_size, void* d_ws, size_t ws_size,
                              hipStream_t stream) {
    const float*     x    = (const float*)d_in[0];
    const long long* ei   = (const long long*)d_in[1];
    const float*     fc_w = (const float*)d_in[2];
    const float*     fc_b = (const float*)d_in[3];
    const float*     w1   = (const float*)d_in[4];
    const float*     b1   = (const float*)d_in[5];
    const float*     g1   = (const float*)d_in[6];
    const float*     be1  = (const float*)d_in[7];
    const float*     w2   = (const float*)d_in[8];
    const float*     b2   = (const float*)d_in[9];
    const float*     t    = (const float*)d_in[10];
    const float*     ng   = (const float*)d_in[11];
    const float*     nb   = (const float*)d_in[12];
    float* out = (float*)d_out;

    const size_t seg = (size_t)NN * HID * sizeof(float);   // 25.6 MB
    unsigned* mkey = (unsigned*)d_ws;
    float*    den  = (float*)((char*)d_ws + seg);
    float*    num  = (float*)((char*)d_ws + 2 * seg);
    const long long* src = ei;
    const long long* dst = ei + EE;

    k_fc<<<NN / 16, 256, 0, stream>>>(x, fc_w, fc_b, out);
    for (int l = 0; l < 3; ++l) {
        k_init<<<(NN * HID + 255) / 256, 256, 0, stream>>>(mkey, den, num);
        k_edge_max<<<(EE * 32) / 256, 256, 0, stream>>>(out + (size_t)l * HID, src, dst,
                                                        t, l, mkey);
        k_edge_sum<<<(EE * 32) / 256, 256, 0, stream>>>(out + (size_t)l * HID, src, dst,
                                                        t, l, mkey, den, num);
        k_node_mlp<<<NN / 16, 256, 0, stream>>>(l, out, w1, b1, g1, be1, w2, b2,
                                                ng, nb, num, den);
    }
}
